// WindowAttention_76424648065817
// MI455X (gfx1250) — matrix operand order repaired and checked by reading, not compiled, not run on hardware
//
#include <hip/hip_runtime.h>
#include <hip/hip_bf16.h>

// ---------------------------------------------------------------------------
// Fused Swin window attention for gfx1250 (MI455X), wave32, WMMA f16->f32.
//   grid = 4096 windows, block = 256 threads (8 waves), 1 window per block.
//   All GEMMs on v_wmma_f32_16x16x32_f16; intermediates in LDS (280KB/WGP).
// ---------------------------------------------------------------------------

#define DIM     512
#define NHEADS  16
#define HDIM    32
#define NTOK    49
#define NPAD    64
#define QKV_N   (3 * DIM)

typedef _Float16 h16;
typedef __attribute__((ext_vector_type(8)))  _Float16 v8h;
typedef __attribute__((ext_vector_type(16))) _Float16 v16h;
typedef __attribute__((ext_vector_type(8)))  float    v8f;

// LDS strides in halves; chosen so every b128 fragment chunk is 16B aligned
// and consecutive rows land on different bank groups.
#define XS_STRIDE 520   // 1040B rows for the 64x512 buffers (Xs, Qs, Ks, Os)
#define PS_STRIDE 72    // 144B rows for per-wave P [64x64]
#define VT_STRIDE 72    // 144B rows for V^T per head [32 x 64]

#define OFF_A 0
#define SZ_A  (8 * NPAD * PS_STRIDE * 2)          // 73728 (holds Xs 66560, then 8 P slices)
#define OFF_Q (OFF_A + SZ_A)
#define SZ_QK (NPAD * XS_STRIDE * 2)              // 66560
#define OFF_K (OFF_Q + SZ_QK)
#define OFF_V (OFF_K + SZ_QK)
#define SZ_V  (NHEADS * HDIM * VT_STRIDE * 2)     // 73728
#define LDS_TOTAL (OFF_V + SZ_V)                  // 280576 bytes <= 320KB

// A-matrix fragment (16x32 f16): lane l holds row (row0 + l&15);
// halves 0..15 = k (l>=16?16:0)+0..15 (contiguous along the row of `base`).
static __device__ __forceinline__ v16h frag_a(const h16* base, int stride,
                                              int row0, int kofs, int lane) {
  const int sub = lane >> 4;
  const h16* p = base + (row0 + (lane & 15)) * stride + kofs + sub * 16;
  v8h lo = *(const v8h*)(p);
  v8h hi = *(const v8h*)(p + 8);
  return __builtin_shufflevector(lo, hi, 0,1,2,3,4,5,6,7,8,9,10,11,12,13,14,15);
}

// B-matrix fragment (32x16 f16): lane l holds column (col0 + l&15);
// halves 0..15 = k (l>=16?16:0)+0..15 (contiguous along the row of `base`).
static __device__ __forceinline__ v16h frag_b(const h16* base, int stride,
                                              int col0, int kofs, int lane) {
  const int sub = lane >> 4;
  const h16* p = base + (col0 + (lane & 15)) * stride + kofs + sub * 16;
  v8h lo = *(const v8h*)(p);
  v8h hi = *(const v8h*)(p + 8);
  return __builtin_shufflevector(lo, hi, 0,1,2,3,4,5,6,7,8,9,10,11,12,13,14,15);
}

static __device__ __forceinline__ v8f wmma_f16(v16h a, v16h b, v8f c) {
  return __builtin_amdgcn_wmma_f32_16x16x32_f16(false, a, false, b,
                                                (short)0, c, false, false);
}

// ---------------------------------------------------------------------------
// Pre-swizzle fp32 weights [512 x ncols] into f16 WMMA B-fragment order:
//   dst[((nt*16+ks)*32+lane)*16 + j] = W[(ks*32 + (lane>>4)*16 + j)*ncols + nt*16 + (lane&15)]
// so the GEMM loop loads one contiguous v16h (32B) per lane per k-step.
// ---------------------------------------------------------------------------
__global__ void swin_swizzle_w(const float* __restrict__ W, h16* __restrict__ dst,
                               int ncols, int total) {
  int t = blockIdx.x * 256 + threadIdx.x;
  if (t >= total) return;
  int j    = t & 15;
  int lane = (t >> 4) & 31;
  int ks   = (t >> 9) & 15;
  int nt   = t >> 13;
  int k = ks * 32 + ((lane >> 4) << 4) + j;
  int n = nt * 16 + (lane & 15);
  dst[t] = (h16)W[(size_t)k * ncols + n];
}

// ---------------------------------------------------------------------------
__global__ __launch_bounds__(256)
void swin_window_attn(const float* __restrict__ x,
                      const float* __restrict__ qkv_b,
                      const float* __restrict__ proj_b,
                      const float* __restrict__ table,     // [169][16]
                      const int*   __restrict__ relidx,    // [49][49]
                      const h16*   __restrict__ wqkv,      // swizzled [512 x 1536]
                      const h16*   __restrict__ wproj,     // swizzled [512 x 512]
                      float* __restrict__ out) {
  extern __shared__ char smem[];
  h16* Xs = (h16*)(smem + OFF_A);               // stage 1-2, then P slices
  h16* Qs = (h16*)(smem + OFF_Q);               // stage 2-3, then O staging
  h16* Ks = (h16*)(smem + OFF_K);
  h16* Vt = (h16*)(smem + OFF_V);               // per-head transposed V

  const int blk  = blockIdx.x;
  const int tid  = threadIdx.x;
  const int lane = tid & 31;
  const int wave = tid >> 5;
  const int kl   = lane & 15;
  const int sub  = lane >> 4;

  // ---------------- stage 1: x window -> LDS f16 (rows 49..63 zero) --------
  for (int i = tid; i < NPAD * (DIM / 4); i += 256) {
    const int r  = i >> 7;              // 128 float4 per row
    const int c4 = (i & 127) << 2;
    float4 v = make_float4(0.f, 0.f, 0.f, 0.f);
    if (r < NTOK)
      v = *(const float4*)(x + ((size_t)blk * NTOK + r) * DIM + c4);
    h16* d = Xs + r * XS_STRIDE + c4;
    d[0] = (h16)v.x; d[1] = (h16)v.y; d[2] = (h16)v.z; d[3] = (h16)v.w;
  }
  __syncthreads();

  // ---------------- stage 2: QKV GEMM [64x512] @ [512x1536] ----------------
  // 96 N-tiles, 12 per wave; K loop = 16 steps of 32.
  #pragma unroll 1
  for (int i = 0; i < 12; ++i) {
    const int nt = wave * 12 + i;
    v8f acc[4] = {};
    for (int ks = 0; ks < 16; ++ks) {
      v16h bf = *(const v16h*)(wqkv + (((size_t)nt * 16 + ks) * 32 + lane) * 16);
      #pragma unroll
      for (int mt = 0; mt < 4; ++mt) {
        v16h af = frag_a(Xs, XS_STRIDE, mt * 16, ks * 32, lane);
        acc[mt] = wmma_f16(af, bf, acc[mt]);
      }
    }
    const int gcol = nt * 16 + kl;
    const float bb = qkv_b[gcol];
    #pragma unroll
    for (int mt = 0; mt < 4; ++mt) {
      #pragma unroll
      for (int j = 0; j < 8; ++j) {
        const int row = mt * 16 + sub * 8 + j;
        const float v = acc[mt][j] + bb;
        if (gcol < DIM) {                                   // Q, pre-scaled
          Qs[row * XS_STRIDE + gcol] = (h16)(v * 0.17677669529663689f);
        } else if (gcol < 2 * DIM) {                        // K, row major
          Ks[row * XS_STRIDE + (gcol - DIM)] = (h16)v;
        } else {                                            // V, transposed/head
          const int f = gcol - 2 * DIM;
          Vt[(f >> 5) * (HDIM * VT_STRIDE) + (f & 31) * VT_STRIDE + row] = (h16)v;
        }
      }
    }
  }
  __syncthreads();

  // ---------------- stage 3: attention, 2 heads per wave -------------------
  h16* Psw = Xs + wave * (NPAD * PS_STRIDE);    // per-wave P slice (Xs is dead)
  v8h oh[2][4][2];                              // O held as f16 in registers

  #pragma unroll 1
  for (int hh = 0; hh < 2; ++hh) {
    const int h = wave * 2 + hh;

    // S = Q_h @ K_h^T : 4x4 tiles, single K step (K = 32 = head dim)
    v16h afr[4], bfr[4];
    #pragma unroll
    for (int mt = 0; mt < 4; ++mt) afr[mt] = frag_a(Qs, XS_STRIDE, mt * 16, h * HDIM, lane);
    #pragma unroll
    for (int nt = 0; nt < 4; ++nt) bfr[nt] = frag_b(Ks, XS_STRIDE, nt * 16, h * HDIM, lane);
    v8f s[4][4];
    #pragma unroll
    for (int mt = 0; mt < 4; ++mt) {
      #pragma unroll
      for (int nt = 0; nt < 4; ++nt) {
        v8f z = {};
        s[mt][nt] = wmma_f16(afr[mt], bfr[nt], z);
      }
    }

    // bias + mask + row softmax (rows live across 16 lanes of a half-wave)
    #pragma unroll
    for (int mt = 0; mt < 4; ++mt) {
      #pragma unroll
      for (int j = 0; j < 8; ++j) {
        const int q = mt * 16 + sub * 8 + j;
        float vals[4];
        #pragma unroll
        for (int nt = 0; nt < 4; ++nt) {
          const int kk = nt * 16 + kl;
          float v = s[mt][nt][j];
          if (q < NTOK && kk < NTOK) v += table[relidx[q * NTOK + kk] * NHEADS + h];
          if (kk >= NTOK) v = -1e30f;
          if (q >= NTOK)  v = 0.0f;     // padded query rows: finite garbage
          vals[nt] = v;
        }
        float m = fmaxf(fmaxf(vals[0], vals[1]), fmaxf(vals[2], vals[3]));
        #pragma unroll
        for (int d = 8; d > 0; d >>= 1) m = fmaxf(m, __shfl_xor(m, d, 32));
        float sum = 0.f;
        #pragma unroll
        for (int nt = 0; nt < 4; ++nt) { vals[nt] = __expf(vals[nt] - m); sum += vals[nt]; }
        #pragma unroll
        for (int d = 8; d > 0; d >>= 1) sum += __shfl_xor(sum, d, 32);
        const float inv = 1.0f / sum;
        #pragma unroll
        for (int nt = 0; nt < 4; ++nt)
          Psw[q * PS_STRIDE + nt * 16 + kl] = (h16)(vals[nt] * inv);
      }
    }

    // O = P @ V_h : [64x64] @ [64x32], K = 64 in two steps
    v8f o[4][2] = {};
    #pragma unroll
    for (int ksp = 0; ksp < 2; ++ksp) {
      v16h pa[4], vb[2];
      #pragma unroll
      for (int mt = 0; mt < 4; ++mt) pa[mt] = frag_a(Psw, PS_STRIDE, mt * 16, ksp * 32, lane);
      #pragma unroll
      for (int nd = 0; nd < 2; ++nd)
        vb[nd] = frag_b(Vt + h * (HDIM * VT_STRIDE), VT_STRIDE, nd * 16, ksp * 32, lane);
      #pragma unroll
      for (int mt = 0; mt < 4; ++mt)
        #pragma unroll
        for (int nd = 0; nd < 2; ++nd)
          o[mt][nd] = wmma_f16(pa[mt], vb[nd], o[mt][nd]);
    }
    #pragma unroll
    for (int mt = 0; mt < 4; ++mt)
      #pragma unroll
      for (int nd = 0; nd < 2; ++nd)
        oh[hh][mt][nd] = __builtin_convertvector(o[mt][nd], v8h);
  }
  __syncthreads();                  // everyone done reading Qs/Ks

  // stage O (f16) into the dead Q region as a row-major [64 x 512] matrix
  h16* Os = Qs;
  #pragma unroll
  for (int hh = 0; hh < 2; ++hh)
    #pragma unroll
    for (int mt = 0; mt < 4; ++mt)
      #pragma unroll
      for (int nd = 0; nd < 2; ++nd)
        #pragma unroll
        for (int j = 0; j < 8; ++j) {
          const int row = mt * 16 + sub * 8 + j;
          const int col = (wave * 2 + hh) * HDIM + nd * 16 + kl;
          Os[row * XS_STRIDE + col] = oh[hh][mt][nd][j];
        }
  __syncthreads();

  // ---------------- stage 4: proj GEMM [64x512] @ [512x512] + bias ---------
  #pragma unroll 1
  for (int nn = 0; nn < 4; ++nn) {
    const int nt = wave * 4 + nn;
    v8f acc[4] = {};
    for (int ks = 0; ks < 16; ++ks) {
      v16h bf = *(const v16h*)(wproj + (((size_t)nt * 16 + ks) * 32 + lane) * 16);
      #pragma unroll
      for (int mt = 0; mt < 4; ++mt) {
        v16h af = frag_a(Os, XS_STRIDE, mt * 16, ks * 32, lane);
        acc[mt] = wmma_f16(af, bf, acc[mt]);
      }
    }
    const int col = nt * 16 + kl;
    const float pb = proj_b[col];
    #pragma unroll
    for (int mt = 0; mt < 4; ++mt)
      #pragma unroll
      for (int j = 0; j < 8; ++j) {
        const int row = mt * 16 + sub * 8 + j;
        if (row < NTOK)
          out[((size_t)blk * NTOK + row) * DIM + col] = acc[mt][j] + pb;
      }
  }
}

// ---------------------------------------------------------------------------
extern "C" void kernel_launch(void* const* d_in, const int* in_sizes, int n_in,
                              void* d_out, int out_size, void* d_ws, size_t ws_size,
                              hipStream_t stream) {
  (void)in_sizes; (void)n_in; (void)out_size; (void)ws_size;
  const float* x       = (const float*)d_in[0];
  const float* qkv_w   = (const float*)d_in[1];
  const float* qkv_b   = (const float*)d_in[2];
  const float* proj_w  = (const float*)d_in[3];
  const float* proj_b  = (const float*)d_in[4];
  const float* table   = (const float*)d_in[5];
  const int*   relidx  = (const int*)  d_in[6];
  float*       out     = (float*)d_out;

  h16* wqkv  = (h16*)d_ws;                       // 512*1536 f16 = 1.5 MB
  h16* wproj = wqkv + (size_t)DIM * QKV_N;       // 512*512  f16 = 0.5 MB

  const int tot_qkv  = DIM * QKV_N;              // 786432
  const int tot_proj = DIM * DIM;                // 262144
  swin_swizzle_w<<<(tot_qkv  + 255) / 256, 256, 0, stream>>>(qkv_w,  wqkv,  QKV_N, tot_qkv);
  swin_swizzle_w<<<(tot_proj + 255) / 256, 256, 0, stream>>>(proj_w, wproj, DIM,   tot_proj);

  swin_window_attn<<<4096, 256, LDS_TOTAL, stream>>>(
      x, qkv_b, proj_b, table, relidx, wqkv, wproj, out);
}
